// AttentionWindow_84722524881145
// MI455X (gfx1250) — matrix operand order repaired and checked by reading, not compiled, not run on hardware
//
#include <hip/hip_runtime.h>
#include <hip/hip_bf16.h>

typedef __attribute__((ext_vector_type(16))) _Float16 v16h;
typedef __attribute__((ext_vector_type(8)))  _Float16 v8h;
typedef __attribute__((ext_vector_type(8)))  float    v8f;

#define BATCH 64
#define SEQ   197
#define EMB   768
#define NH    12
#define HD    64
#define MTOT  (BATCH*SEQ)        // 12608 = 394 * 32
#define QSCALE 0.125f            // 64^-0.5

union U16h { v16h v; v8h h[2]; _Float16 e[16]; };

__device__ __forceinline__ int imin(int a, int b) { return a < b ? a : b; }

// ---------------------------------------------------------------- f32 -> f16
__global__ void k_f32_to_f16(const float* __restrict__ s, _Float16* __restrict__ d, int n) {
    int i = blockIdx.x * blockDim.x + threadIdx.x;
    if (i < n) d[i] = (_Float16)s[i];
}

// ---------------------------------------------------------------- QKV GEMM (32x64 block / wave)
// out = x @ qkv_w.T + [q_bias, 0, v_bias]; scatter into q/k/v (B,H,N,HD), q pre-scaled.
// N-block of 64 == exactly one head of one of {q,k,v}.
__global__ __launch_bounds__(32) void k_qkv_gemm(
    const _Float16* __restrict__ xh, const _Float16* __restrict__ wh,
    const float* __restrict__ q_bias, const float* __restrict__ v_bias,
    _Float16* __restrict__ qh, _Float16* __restrict__ kh, _Float16* __restrict__ vh)
{
    const int lane = threadIdx.x;
    const int hi   = lane >> 4;      // lane half (0/1)
    const int l15  = lane & 15;
    const int tm = blockIdx.y;       // 0..393  (M block of 32)
    const int tn = blockIdx.x;       // 0..35   (N block of 64)
    const _Float16* arow[2];
    arow[0] = xh + (size_t)(tm * 32 + l15) * EMB;
    arow[1] = arow[0] + (size_t)16 * EMB;
    const _Float16* brow[4];
#pragma unroll
    for (int s = 0; s < 4; ++s)
        brow[s] = wh + (size_t)(tn * 64 + s * 16 + l15) * EMB;   // qkv_w[n][k]

    v8f acc[2][4];
    const v8f vzero = {};
#pragma unroll
    for (int mi = 0; mi < 2; ++mi)
#pragma unroll
        for (int s = 0; s < 4; ++s) acc[mi][s] = vzero;

    for (int k0 = 0; k0 < EMB; k0 += 32) {
        U16h a[2], bf[4];
        const int ka = k0 + hi * 8;          // A: elems0-7=K ka.., elems8-15=K ka+16..
        const int kb = k0 + hi * 8;          // B: elems0-7=K kb.., elems8-15=K kb+16..
#pragma unroll
        for (int mi = 0; mi < 2; ++mi) {
            a[mi].h[0] = *(const v8h*)(arow[mi] + ka);
            a[mi].h[1] = *(const v8h*)(arow[mi] + ka + 16);
        }
#pragma unroll
        for (int s = 0; s < 4; ++s) {
            bf[s].h[0] = *(const v8h*)(brow[s] + kb);
            bf[s].h[1] = *(const v8h*)(brow[s] + kb + 16);
        }
#pragma unroll
        for (int mi = 0; mi < 2; ++mi)
#pragma unroll
            for (int s = 0; s < 4; ++s)
                acc[mi][s] = __builtin_amdgcn_wmma_f32_16x16x32_f16(
                    false, a[mi].v, false, bf[s].v, (short)0, acc[mi][s], false, false);
    }

    const int sec  = tn / NH;        // 0=q 1=k 2=v
    const int head = tn % NH;
    _Float16* dst = (sec == 0) ? qh : (sec == 1) ? kh : vh;
    const float osc = (sec == 0) ? QSCALE : 1.0f;
#pragma unroll
    for (int s = 0; s < 4; ++s) {
        const int d = s * 16 + l15;
        float bias = (sec == 0) ? q_bias[head * HD + d]
                   : (sec == 2) ? v_bias[head * HD + d] : 0.0f;
#pragma unroll
        for (int mi = 0; mi < 2; ++mi)
            for (int r = 0; r < 8; ++r) {    // D: VGPR r -> M = r + hi*8
                int mm = tm * 32 + mi * 16 + r + hi * 8;
                int bi = mm / SEQ, pos = mm % SEQ;
                float val = (acc[mi][s][r] + bias) * osc;
                dst[(((size_t)bi * NH + head) * SEQ + pos) * HD + d] = (_Float16)val;
            }
    }
}

// ---------------------------------------------------------------- fused flash attention
// one wave per (b, h, 16-query tile); 7 key tiles of 32 (keys >= 197 masked)
// V tiles staged memory->LDS with GLOBAL_LOAD_ASYNC_TO_LDS_B128 (ASYNCcnt),
// drained under the QK^T WMMAs + softmax.
__global__ __launch_bounds__(32) void k_attn(
    const _Float16* __restrict__ qh, const _Float16* __restrict__ kh,
    const _Float16* __restrict__ vh, const float* __restrict__ rpb,
    const int* __restrict__ relidx, _Float16* __restrict__ aoh)
{
    __shared__ __align__(16) _Float16 pbuf[16 * 32];   // P tile, row-major (q x key)
    __shared__ __align__(16) _Float16 vbuf[32 * HD];   // V tile (key x d)
    const int lane = threadIdx.x;
    const int hi = lane >> 4, l15 = lane & 15;
    const int qt = blockIdx.x % 13;
    const int bh = blockIdx.x / 13;
    const int b  = bh / NH, h = bh % NH;

    // Q fragments: A 16x32 for head-dim chunks [0,32) and [32,64)
    const int qr = imin(qt * 16 + l15, SEQ - 1);
    const _Float16* qptr = qh + ((size_t)bh * SEQ + qr) * HD;
    U16h aq[2];
    for (int c2 = 0; c2 < 2; ++c2) {
        int ka = c2 * 32 + hi * 8;
        aq[c2].h[0] = *(const v8h*)(qptr + ka);
        aq[c2].h[1] = *(const v8h*)(qptr + ka + 16);
    }
    v8f o0 = {}, o1 = {}, o2 = {}, o3 = {};
    float mrun[8], lrun[8];
    for (int r = 0; r < 8; ++r) { mrun[r] = -3.0e38f; lrun[r] = 0.0f; }

    // lane-private LDS destination (group-relative offset = low 32 bits of flat addr)
    const unsigned int vbuf_lds = (unsigned int)(uintptr_t)(vbuf + lane * HD);

    for (int kt = 0; kt < 7; ++kt) {
        // async-stage V tile: lane -> one key row (128 B = 8 x b128), no VGPR transit
        {
            int key = kt * 32 + lane;
            const _Float16* src = vh + ((size_t)bh * SEQ + imin(key, SEQ - 1)) * HD;
            unsigned long long ga = (unsigned long long)(uintptr_t)src;
#define V_ASYNC(ofs) asm volatile("global_load_async_to_lds_b128 %0, %1, off offset:" #ofs \
                                  :: "v"(vbuf_lds), "v"(ga) : "memory")
            V_ASYNC(0);  V_ASYNC(16); V_ASYNC(32); V_ASYNC(48);
            V_ASYNC(64); V_ASYNC(80); V_ASYNC(96); V_ASYNC(112);
#undef V_ASYNC
        }
        // scores S = Q K^T for two 16-key subtiles (hides async-copy latency)
        v8f s[2];
        for (int j = 0; j < 2; ++j) {
            int key = kt * 32 + j * 16 + l15;
            const _Float16* kp = kh + ((size_t)bh * SEQ + imin(key, SEQ - 1)) * HD;
            v8f sj = {};
            for (int c2 = 0; c2 < 2; ++c2) {
                U16h bb;
                int kb = c2 * 32 + hi * 8;
                bb.h[0] = *(const v8h*)(kp + kb);
                bb.h[1] = *(const v8h*)(kp + kb + 16);
                sj = __builtin_amdgcn_wmma_f32_16x16x32_f16(false, aq[c2].v, false, bb.v,
                                                            (short)0, sj, false, false);
            }
            s[j] = sj;
        }
        // relative position bias gather + key masking
        for (int j = 0; j < 2; ++j) {
            int key = kt * 32 + j * 16 + l15;
            if (key < SEQ) {
                for (int r = 0; r < 8; ++r) {
                    int q = imin(qt * 16 + r + hi * 8, SEQ - 1);
                    s[j][r] += rpb[relidx[q * SEQ + key] * NH + h];
                }
            } else {
                for (int r = 0; r < 8; ++r) s[j][r] = -3.0e38f;
            }
        }
        // online softmax (row = 16 lanes of one half, two subtiles)
        float p0a[8], p1a[8], rsc[8];
        for (int r = 0; r < 8; ++r) {
            float rm = fmaxf(s[0][r], s[1][r]);
            for (int msk = 8; msk >= 1; msk >>= 1) rm = fmaxf(rm, __shfl_xor(rm, msk, 32));
            float newm = fmaxf(mrun[r], rm);
            float sc = __expf(mrun[r] - newm);
            float p0 = __expf(s[0][r] - newm);
            float p1 = __expf(s[1][r] - newm);
            float rs = p0 + p1;
            for (int msk = 8; msk >= 1; msk >>= 1) rs += __shfl_xor(rs, msk, 32);
            lrun[r] = lrun[r] * sc + rs;
            mrun[r] = newm;
            rsc[r] = sc; p0a[r] = p0; p1a[r] = p1;
        }
        for (int r = 0; r < 8; ++r) {
            float sc = rsc[r];
            o0[r] *= sc; o1[r] *= sc; o2[r] *= sc; o3[r] *= sc;
        }
        // C-layout -> A-layout transpose of P through LDS
        for (int r = 0; r < 8; ++r) {
            int M = r + hi * 8;
            pbuf[M * 32 + l15]      = (_Float16)p0a[r];
            pbuf[M * 32 + 16 + l15] = (_Float16)p1a[r];
        }
        // drain async V copies, then make LDS visible across lanes
        asm volatile("s_wait_asynccnt 0x0" ::: "memory");
        __syncthreads();
        U16h ap;
        {
            const _Float16* pr = pbuf + l15 * 32;
            int ka = hi * 16;
            ap.h[0] = *(const v8h*)(pr + ka);
            ap.h[1] = *(const v8h*)(pr + ka + 8);
        }
        // O += P V : 4 head-dim subtiles of 16
        const int kb0 = hi * 16;
        for (int t = 0; t < 4; ++t) {
            U16h bv;
            int d = t * 16 + l15;
            for (int j = 0; j < 16; ++j) bv.e[j] = vbuf[(kb0 + j) * HD + d];
            v8f* op = (t == 0) ? &o0 : (t == 1) ? &o1 : (t == 2) ? &o2 : &o3;
            *op = __builtin_amdgcn_wmma_f32_16x16x32_f16(false, ap.v, false, bv.v,
                                                         (short)0, *op, false, false);
        }
        __syncthreads();
    }
    // finalize: divide by row sums, store (B, N, H*64) f16
    for (int r = 0; r < 8; ++r) {
        int q = qt * 16 + r + hi * 8;
        if (q >= SEQ) continue;
        float inv = 1.0f / lrun[r];
        size_t base = ((size_t)b * SEQ + q) * EMB + h * HD;
        aoh[base + 0  + l15] = (_Float16)(o0[r] * inv);
        aoh[base + 16 + l15] = (_Float16)(o1[r] * inv);
        aoh[base + 32 + l15] = (_Float16)(o2[r] * inv);
        aoh[base + 48 + l15] = (_Float16)(o3[r] * inv);
    }
}

// ---------------------------------------------------------------- output projection (32x64 block / wave)
__global__ __launch_bounds__(32) void k_proj_gemm(
    const _Float16* __restrict__ ah, const _Float16* __restrict__ wh,
    const float* __restrict__ proj_b, float* __restrict__ out)
{
    const int lane = threadIdx.x;
    const int hi = lane >> 4, l15 = lane & 15;
    const int tm = blockIdx.y;       // 0..393
    const int tn = blockIdx.x;       // 0..11
    const _Float16* arow[2];
    arow[0] = ah + (size_t)(tm * 32 + l15) * EMB;
    arow[1] = arow[0] + (size_t)16 * EMB;
    const _Float16* brow[4];
#pragma unroll
    for (int s = 0; s < 4; ++s)
        brow[s] = wh + (size_t)(tn * 64 + s * 16 + l15) * EMB;   // proj_w[n][k]

    v8f acc[2][4];
    const v8f vzero = {};
#pragma unroll
    for (int mi = 0; mi < 2; ++mi)
#pragma unroll
        for (int s = 0; s < 4; ++s) acc[mi][s] = vzero;

    for (int k0 = 0; k0 < EMB; k0 += 32) {
        U16h a[2], bf[4];
        const int ka = k0 + hi * 8;
        const int kb = k0 + hi * 8;
#pragma unroll
        for (int mi = 0; mi < 2; ++mi) {
            a[mi].h[0] = *(const v8h*)(arow[mi] + ka);
            a[mi].h[1] = *(const v8h*)(arow[mi] + ka + 16);
        }
#pragma unroll
        for (int s = 0; s < 4; ++s) {
            bf[s].h[0] = *(const v8h*)(brow[s] + kb);
            bf[s].h[1] = *(const v8h*)(brow[s] + kb + 16);
        }
#pragma unroll
        for (int mi = 0; mi < 2; ++mi)
#pragma unroll
            for (int s = 0; s < 4; ++s)
                acc[mi][s] = __builtin_amdgcn_wmma_f32_16x16x32_f16(
                    false, a[mi].v, false, bf[s].v, (short)0, acc[mi][s], false, false);
    }

#pragma unroll
    for (int s = 0; s < 4; ++s) {
        int n = tn * 64 + s * 16 + l15;
        float bias = proj_b[n];
#pragma unroll
        for (int mi = 0; mi < 2; ++mi)
            for (int r = 0; r < 8; ++r) {
                int mm = tm * 32 + mi * 16 + r + hi * 8;
                out[(size_t)mm * EMB + n] = acc[mi][s][r] + bias;
            }
    }
}

// ---------------------------------------------------------------- launch
extern "C" void kernel_launch(void* const* d_in, const int* in_sizes, int n_in,
                              void* d_out, int out_size, void* d_ws, size_t ws_size,
                              hipStream_t stream) {
    (void)in_sizes; (void)n_in; (void)out_size; (void)ws_size;
    const float* x      = (const float*)d_in[0];
    const float* qkv_w  = (const float*)d_in[1];
    const float* q_bias = (const float*)d_in[2];
    const float* v_bias = (const float*)d_in[3];
    const float* rpb    = (const float*)d_in[4];
    const float* proj_w = (const float*)d_in[5];
    const float* proj_b = (const float*)d_in[6];
    const int*   relidx = (const int*)d_in[7];
    float* out = (float*)d_out;

    const size_t nX     = (size_t)MTOT * EMB;            // 9,682,944
    const size_t nWqkv  = (size_t)3 * EMB * EMB;         // 1,769,472
    const size_t nWproj = (size_t)EMB * EMB;             //   589,824
    const size_t nHead  = (size_t)BATCH * NH * SEQ * HD; // == nX

    _Float16* xh     = (_Float16*)d_ws;
    _Float16* wqkvh  = xh + nX;
    _Float16* wprojh = wqkvh + nWqkv;
    _Float16* qh     = wprojh + nWproj;
    _Float16* kh2    = qh + nHead;
    _Float16* vh2    = kh2 + nHead;
    _Float16* aoh    = vh2 + nHead;

    const int thr = 256;
    k_f32_to_f16<<<(int)((nX + thr - 1) / thr),     thr, 0, stream>>>(x,      xh,     (int)nX);
    k_f32_to_f16<<<(int)((nWqkv + thr - 1) / thr),  thr, 0, stream>>>(qkv_w,  wqkvh,  (int)nWqkv);
    k_f32_to_f16<<<(int)((nWproj + thr - 1) / thr), thr, 0, stream>>>(proj_w, wprojh, (int)nWproj);

    dim3 g1(3 * EMB / 64, MTOT / 32);   // 36 x 394 blocks (32x64 tile each)
    k_qkv_gemm<<<g1, 32, 0, stream>>>(xh, wqkvh, q_bias, v_bias, qh, kh2, vh2);

    k_attn<<<BATCH * NH * 13, 32, 0, stream>>>(qh, kh2, vh2, rpb, relidx, aoh);

    dim3 g3(EMB / 64, MTOT / 32);       // 12 x 394 blocks
    k_proj_gemm<<<g3, 32, 0, stream>>>(aoh, wprojh, proj_b, out);
}
